// RNN_54915451846837
// MI455X (gfx1250) — compile-verified, matrix operand order checked by reading, not run on hardware
//
#include <hip/hip_runtime.h>
#include <hip/hip_bf16.h>

typedef __attribute__((ext_vector_type(16))) _Float16 v16h;
typedef __attribute__((ext_vector_type(8)))  float    v8f;

#define T_STEPS 140
#define H_DIM   64
#define B_DIM   8192

__device__ __forceinline__ float fast_tanh(float x) {
#if __has_builtin(__builtin_amdgcn_tanhf)
    return __builtin_amdgcn_tanhf(x);                 // v_tanh_f32 (gfx1250 trans op)
#elif __has_builtin(__builtin_amdgcn_tanh_f32)
    return __builtin_amdgcn_tanh_f32(x);
#else
    // tanh(x) = 1 - 2/(exp2(x*2*log2e)+1)
    float e = __builtin_amdgcn_exp2f(x * 2.8853900817779268f);
    return 1.0f - 2.0f * __builtin_amdgcn_rcpf(e + 1.0f);
#endif
}

// One wave handles 16 batch columns. Transposed recurrence:
//   h_new^T(16 feat x 16 batch tiles) = W(A, 16x32 f16) x h^T(B, 32x16 f16) + C(f32)
// D->B repack for the next step is lane-local (no cross-lane traffic).
__global__ __launch_bounds__(128) void rnn_wmma_kernel(
    const float* __restrict__ X,    const float* __restrict__ Hst,
    const float* __restrict__ Wih0, const float* __restrict__ Whh0,
    const float* __restrict__ bih0, const float* __restrict__ bhh0,
    const float* __restrict__ Wih1, const float* __restrict__ Whh1,
    const float* __restrict__ bih1, const float* __restrict__ bhh1,
    const float* __restrict__ Wout, const float* __restrict__ bout,
    float* __restrict__ out, float* __restrict__ hfin)
{
    // LDS: 3 weight matrices pre-swizzled into WMMA A-operand layout.
    // Per matrix: 8 blocks (4 M-tiles x 2 K-tiles), each block = 32 lanes x 16 halfs.
    __shared__ __align__(32) _Float16 sW[3 * 4096];

    for (int idx = threadIdx.x; idx < 4096; idx += 128) {
        int f  = idx >> 6;           // row (out feature) = M
        int k  = idx & 63;           // col (in feature)  = K
        int mt = f >> 4, l0 = f & 15;
        int kt = k >> 5, kk = k & 31;
        int grp  = kk >> 3;                        // 0..3
        int hi   = grp & 1;                        // lane half
        int e    = (kk & 7) + ((grp >> 1) << 3);   // half index 0..15
        int lane = l0 + (hi << 4);
        int off  = (mt * 2 + kt) * 512 + lane * 16 + e;
        sW[off]        = (_Float16)Whh0[idx];
        sW[4096 + off] = (_Float16)Wih1[idx];
        sW[8192 + off] = (_Float16)Whh1[idx];
    }
    __syncthreads();

    const int lane  = threadIdx.x & 31;
    const int wid   = threadIdx.x >> 5;
    const int hiL   = (lane >> 4) & 1;
    const int col   = lane & 15;
    const int batch = (blockIdx.x * 4 + wid) * 16 + col;

    // Per-lane tile-layout constants (feature f = 16n + 8*hiL + m)
    float bias0r[4][8], wih0r[4][8], bias1r[4][8], woutr[4][8];
#pragma unroll
    for (int n = 0; n < 4; ++n)
#pragma unroll
        for (int m = 0; m < 8; ++m) {
            int f = 16 * n + hiL * 8 + m;
            bias0r[n][m] = bih0[f] + bhh0[f];
            bias1r[n][m] = bih1[f] + bhh1[f];
            wih0r[n][m]  = Wih0[f];
            woutr[n][m]  = Wout[f];
        }
    const float bo = bout[0];

    // Initial hidden state as f16 B operands (K = feature, N = batch column).
    v16h Bh0[2], Bh1[2];
#pragma unroll
    for (int kt = 0; kt < 2; ++kt)
#pragma unroll
        for (int e = 0; e < 16; ++e) {
            int K = 32 * kt + ((e < 8) ? (hiL * 8 + e) : (16 + hiL * 8 + (e - 8)));
            Bh0[kt][e] = (_Float16)Hst[batch * H_DIM + K];
            Bh1[kt][e] = (_Float16)Hst[(B_DIM + batch) * H_DIM + K];
        }

    const _Float16* sW0 = sW;
    const _Float16* sW1 = sW + 4096;
    const _Float16* sW2 = sW + 8192;

    for (int t = 0; t < T_STEPS; ++t) {
        float xv = X[batch * T_STEPS + t];

        // ----- layer 0: C = b_ih0 + b_hh0 + x*W_ih0 ; D = W_hh0 * h0^T + C -----
        v8f acc[4];
#pragma unroll
        for (int n = 0; n < 4; ++n)
#pragma unroll
            for (int m = 0; m < 8; ++m)
                acc[n][m] = fmaf(wih0r[n][m], xv, bias0r[n][m]);

#pragma unroll
        for (int n = 0; n < 4; ++n)
#pragma unroll
            for (int kt = 0; kt < 2; ++kt) {
                const v16h a = *reinterpret_cast<const v16h*>(sW0 + (n * 2 + kt) * 512 + lane * 16);
                acc[n] = __builtin_amdgcn_wmma_f32_16x16x32_f16(
                    false, a, false, Bh0[kt], (short)0, acc[n], false, false);
            }

        // tanh + lane-local repack into next-step B operand
        v16h Bn0[2];
#pragma unroll
        for (int n = 0; n < 4; ++n)
#pragma unroll
            for (int m = 0; m < 8; ++m) {
                float v = fast_tanh(acc[n][m]);
                Bn0[n >> 1][(n & 1) * 8 + m] = (_Float16)v;
            }

        // ----- layer 1: D = W_ih1 * h0n^T + W_hh1 * h1^T + (b_ih1+b_hh1) -----
#pragma unroll
        for (int n = 0; n < 4; ++n)
#pragma unroll
            for (int m = 0; m < 8; ++m)
                acc[n][m] = bias1r[n][m];

#pragma unroll
        for (int n = 0; n < 4; ++n)
#pragma unroll
            for (int kt = 0; kt < 2; ++kt) {
                const v16h a = *reinterpret_cast<const v16h*>(sW1 + (n * 2 + kt) * 512 + lane * 16);
                acc[n] = __builtin_amdgcn_wmma_f32_16x16x32_f16(
                    false, a, false, Bn0[kt], (short)0, acc[n], false, false);
            }
#pragma unroll
        for (int n = 0; n < 4; ++n)
#pragma unroll
            for (int kt = 0; kt < 2; ++kt) {
                const v16h a = *reinterpret_cast<const v16h*>(sW2 + (n * 2 + kt) * 512 + lane * 16);
                acc[n] = __builtin_amdgcn_wmma_f32_16x16x32_f16(
                    false, a, false, Bh1[kt], (short)0, acc[n], false, false);
            }

        // tanh, fused output projection, repack h1
        float part = 0.0f;
        v16h Bn1[2];
#pragma unroll
        for (int n = 0; n < 4; ++n)
#pragma unroll
            for (int m = 0; m < 8; ++m) {
                float v = fast_tanh(acc[n][m]);
                part = fmaf(woutr[n][m], v, part);
                Bn1[n >> 1][(n & 1) * 8 + m] = (_Float16)v;
            }
        Bh0[0] = Bn0[0]; Bh0[1] = Bn0[1];
        Bh1[0] = Bn1[0]; Bh1[1] = Bn1[1];

        part += __shfl_xor(part, 16);           // combine the two feature halves
        if (!hiL)
            out[batch * T_STEPS + t] = part + bo;
    }

    // Final hidden state: unpack resident f16 B operands (each lane covers a
    // disjoint set of features K).
#pragma unroll
    for (int kt = 0; kt < 2; ++kt)
#pragma unroll
        for (int e = 0; e < 16; ++e) {
            int K = 32 * kt + ((e < 8) ? (hiL * 8 + e) : (16 + hiL * 8 + (e - 8)));
            hfin[batch * H_DIM + K]           = (float)Bh0[kt][e];
            hfin[(B_DIM + batch) * H_DIM + K] = (float)Bh1[kt][e];
        }
}

extern "C" void kernel_launch(void* const* d_in, const int* in_sizes, int n_in,
                              void* d_out, int out_size, void* d_ws, size_t ws_size,
                              hipStream_t stream) {
    const float* X    = (const float*)d_in[0];
    const float* Hst  = (const float*)d_in[1];
    const float* Wih0 = (const float*)d_in[2];
    const float* Whh0 = (const float*)d_in[3];
    const float* bih0 = (const float*)d_in[4];
    const float* bhh0 = (const float*)d_in[5];
    const float* Wih1 = (const float*)d_in[6];
    const float* Whh1 = (const float*)d_in[7];
    const float* bih1 = (const float*)d_in[8];
    const float* bhh1 = (const float*)d_in[9];
    const float* Wout = (const float*)d_in[10];
    const float* bout = (const float*)d_in[11];

    float* out  = (float*)d_out;                    // [B, T, 1]
    float* hfin = (float*)d_out + (size_t)B_DIM * T_STEPS;  // [2, B, H]

    // 8192 batch / (4 waves * 16 cols) = 128 blocks of 128 threads (4 wave32)
    rnn_wmma_kernel<<<B_DIM / 64, 128, 0, stream>>>(
        X, Hst, Wih0, Whh0, bih0, bhh0, Wih1, Whh1, bih1, bhh1, Wout, bout,
        out, hfin);
}
